// HGTBackbone_42494406427178
// MI455X (gfx1250) — matrix operand order repaired and checked by reading, not compiled, not run on hardware
//
#include <hip/hip_runtime.h>

// ---------------------------------------------------------------------------
// HGT backbone forward for MI455X (gfx1250), wave32.
// - All dense GEMMs + edge transforms: v_wmma_f32_16x16x32_f16 (f32 accum)
// - All LDS staging: global_load_async_to_lds_b128 (ASYNCcnt), no VGPR bounce
// - GEMM: double-buffered LDS, DMA for tile i+1 overlapped with WMMA on tile i
//   (uniform 6 async issues/wave/tile -> s_wait_asynccnt 6)
// - Weights/relations pre-transposed to f16 [N][K] once per launch
// ---------------------------------------------------------------------------

typedef __attribute__((ext_vector_type(16))) _Float16 v16h;
typedef __attribute__((ext_vector_type(8)))  float    v8f;

union HV { v16h v; uint4 q[2]; };

#define DEV __device__ __forceinline__

DEV float gelu_exact(float x) {
    return 0.5f * x * (1.0f + erff(x * 0.70710678118654752f));
}

// monotonic encoding of float for unsigned atomicMax (order preserving)
DEV unsigned enc_f32(float x) {
    unsigned u = __float_as_uint(x);
    return (u & 0x80000000u) ? ~u : (u | 0x80000000u);
}
DEV float dec_f32(unsigned e) {
    return __uint_as_float((e & 0x80000000u) ? (e ^ 0x80000000u) : ~e);
}

// CDNA5 async memory->LDS copy, 16 bytes per active lane. LDS destination is
// the wave-relative LDS byte offset (low 32 bits of the generic pointer).
DEV void async_copy16(const void* gptr, void* lptr) {
    unsigned lds = (unsigned)(unsigned long long)lptr;
    asm volatile("global_load_async_to_lds_b128 %0, %1, off"
                 :: "v"(lds), "v"((unsigned long long)gptr)
                 : "memory");
}
DEV void async_wait0()   { asm volatile("s_wait_asynccnt 0" ::: "memory"); }
DEV void async_wait_le6(){ asm volatile("s_wait_asynccnt 6" ::: "memory"); }

// ---------------------------------------------------------------------------
// Elementwise helpers
// ---------------------------------------------------------------------------
__global__ void gelu_f32_to_f16_kernel(const float* __restrict__ s, _Float16* __restrict__ d, size_t n) {
    size_t i = (size_t)blockIdx.x * blockDim.x + threadIdx.x;
    size_t st = (size_t)gridDim.x * blockDim.x;
    for (; i < n; i += st) d[i] = (_Float16)gelu_exact(s[i]);
}

__global__ void fill_u32_kernel(unsigned* __restrict__ p, unsigned v, size_t n) {
    size_t i = (size_t)blockIdx.x * blockDim.x + threadIdx.x;
    size_t st = (size_t)gridDim.x * blockDim.x;
    for (; i < n; i += st) p[i] = v;
}

__global__ void bcast_tweet_kernel(const float* __restrict__ tv, float* __restrict__ out, size_t n) {
    size_t i = (size_t)blockIdx.x * blockDim.x + threadIdx.x;
    size_t st = (size_t)gridDim.x * blockDim.x;
    for (; i < n; i += st) out[i] = tv[i & 255];
}

// ---------------------------------------------------------------------------
// Tiled transpose + f32->f16: dst[s][n*K + k] = (f16)src[s][k*N + n]
// grid = (N/32, K/32, slices), block = (32, 8). K, N multiples of 32.
// ---------------------------------------------------------------------------
__global__ __launch_bounds__(256) void transpose_f32_to_f16_kernel(
    const float* __restrict__ src, _Float16* __restrict__ dst, int K, int N)
{
    __shared__ float t[32][33];
    const float* sp = src + (size_t)blockIdx.z * K * N;
    _Float16*    dp = dst + (size_t)blockIdx.z * K * N;
    int kb = blockIdx.y * 32, nb = blockIdx.x * 32;
    int tx = threadIdx.x, ty = threadIdx.y;
#pragma unroll
    for (int i = 0; i < 4; ++i)
        t[ty + i * 8][tx] = sp[(size_t)(kb + ty + i * 8) * N + nb + tx];
    __syncthreads();
#pragma unroll
    for (int i = 0; i < 4; ++i)
        dp[(size_t)(nb + ty + i * 8) * K + kb + tx] = (_Float16)t[tx][ty + i * 8];
}

// ---------------------------------------------------------------------------
// LayerNorm over D=256, one 256-thread block per row; writes f32 and/or f16.
// ---------------------------------------------------------------------------
__global__ __launch_bounds__(256) void ln_kernel(
    const float* __restrict__ x, const float* __restrict__ g, const float* __restrict__ b,
    float* __restrict__ y32, _Float16* __restrict__ y16, int Nrows)
{
    int row = blockIdx.x;
    int tid = threadIdx.x;
    if (row >= Nrows) return;
    __shared__ float red[256];
    float v = x[(size_t)row * 256 + tid];
    red[tid] = v;
    __syncthreads();
    for (int s = 128; s > 0; s >>= 1) { if (tid < s) red[tid] += red[tid + s]; __syncthreads(); }
    float mu = red[0] * (1.0f / 256.0f);
    __syncthreads();
    float d = v - mu;
    red[tid] = d * d;
    __syncthreads();
    for (int s = 128; s > 0; s >>= 1) { if (tid < s) red[tid] += red[tid + s]; __syncthreads(); }
    float var = red[0] * (1.0f / 256.0f);
    float o = d * rsqrtf(var + 1e-5f) * g[tid] + b[tid];
    if (y32) y32[(size_t)row * 256 + tid] = o;
    if (y16) y16[(size_t)row * 256 + tid] = (_Float16)o;
}

// ---------------------------------------------------------------------------
// WMMA GEMM: C[M,N] = epilogue(A[M,K](f16) @ B[K,N] + bias), B given as
// BT[N][K] (pre-transposed f16). Block tile 64x128, 8 waves, BK=64.
// Double-buffered async DMA pipeline: stage tile i+1 while computing tile i.
// modes: 0 plain, 1 gelu, 2 skip-blend (res + beta*o + (1-beta)*aux), 3 res+o
// ---------------------------------------------------------------------------
#define BM 64
#define BN 128
#define BK 64

__global__ __launch_bounds__(256) void gemm_kernel(
    const _Float16* __restrict__ A, const _Float16* __restrict__ BT,
    const float* __restrict__ bias,
    float* __restrict__ C32, int ldc32,
    _Float16* __restrict__ C16, int ldc16,
    const float* __restrict__ res, const float* __restrict__ aux,
    const float* __restrict__ skipP,
    int M, int N, int K, int mode)
{
    __shared__ _Float16 sA[2][BM][BK + 8];   // [64][72], row = 144B (16B aligned)
    __shared__ _Float16 sB[2][BN][BK + 8];   // [128][72], row n = K-slice
    int tid = threadIdx.x, lane = tid & 31, wave = tid >> 5;
    int m0 = blockIdx.y * BM, n0 = blockIdx.x * BN;
    int wm = (wave & 3) * 16;
    int wn = (wave >> 2) * 64;

    v8f acc[4];
#pragma unroll
    for (int j = 0; j < 4; ++j)
#pragma unroll
        for (int r = 0; r < 8; ++r) acc[j][r] = 0.0f;

    // per-thread staging coordinates; clamp A row so every wave issues exactly
    // 6 async loads per tile (required for counted s_wait_asynccnt pipelining).
    int ar = tid >> 2, ac = (tid & 3) * 16;   // A: 64 rows x 64 halves
    int bn = tid >> 1, bc = (tid & 1) * 32;   // B: 128 rows x 64 halves
    int gr = m0 + ar; if (gr >= M) gr = M - 1;

    auto stage = [&](int ti, int bi) {
        int k0 = ti * BK;
        const _Float16* ap = A + (size_t)gr * K + k0 + ac;
        async_copy16(ap,     &sA[bi][ar][ac]);
        async_copy16(ap + 8, &sA[bi][ar][ac + 8]);
        const _Float16* bp = BT + (size_t)(n0 + bn) * K + k0 + bc;
        async_copy16(bp,      &sB[bi][bn][bc]);
        async_copy16(bp + 8,  &sB[bi][bn][bc + 8]);
        async_copy16(bp + 16, &sB[bi][bn][bc + 16]);
        async_copy16(bp + 24, &sB[bi][bn][bc + 24]);
    };

    int T = K / BK;
    stage(0, 0);
    for (int i = 0; i < T; ++i) {
        int cur = i & 1;
        if (i + 1 < T) { stage(i + 1, cur ^ 1); async_wait_le6(); }
        else           { async_wait0(); }
        __syncthreads();

#pragma unroll
        for (int ks = 0; ks < 2; ++ks) {
            int kb = ks * 32;
            // A frag (16x32 ISA layout): M=lane%16, K = kb + {g*8..+7, 16+g*8..+7}
            HV af;
            int row = wm + (lane & 15), g = lane >> 4;
            af.q[0] = *(const uint4*)&sA[cur][row][kb + g * 8];
            af.q[1] = *(const uint4*)&sA[cur][row][kb + 16 + g * 8];
#pragma unroll
            for (int j = 0; j < 4; ++j) {
                // B frag (32x16): N=lane%16, K = kb + {g*8..+7, 16+g*8..+7}
                HV bf;
                int col = wn + j * 16 + (lane & 15);
                bf.q[0] = *(const uint4*)&sB[cur][col][kb + g * 8];
                bf.q[1] = *(const uint4*)&sB[cur][col][kb + 16 + g * 8];
                acc[j] = __builtin_amdgcn_wmma_f32_16x16x32_f16(
                    false, af.v, false, bf.v, (short)0, acc[j], false, false);
            }
        }
        __syncthreads();   // all waves done reading buf before it is re-staged
    }

    // epilogue: C/D layout: VGPR r -> M = 8*(lane/16)+r, N = lane%16
    float beta = 0.0f;
    if (mode == 2) beta = 1.0f / (1.0f + __expf(-skipP[0]));
    int rbase = m0 + wm + (lane >> 4) * 8;
    int cloc = lane & 15;
#pragma unroll
    for (int j = 0; j < 4; ++j) {
        int gn = n0 + wn + j * 16 + cloc;
        float bv = bias ? bias[gn] : 0.0f;
#pragma unroll
        for (int r = 0; r < 8; ++r) {
            int gm = rbase + r;
            if (gm >= M) continue;
            float c = acc[j][r] + bv;
            if (mode == 1) c = gelu_exact(c);
            else if (mode == 2) c = res[(size_t)gm * ldc32 + gn] + beta * c
                                  + (1.0f - beta) * aux[(size_t)gm * ldc32 + gn];
            else if (mode == 3) c = res[(size_t)gm * ldc32 + gn] + c;
            if (C32) C32[(size_t)gm * ldc32 + gn] = c;
            if (C16) C16[(size_t)gm * ldc16 + gn] = (_Float16)c;
        }
    }
}

// ---------------------------------------------------------------------------
// Edge attention: per block of 64 edges, per head h:
//   k_rel = kk[src] @ a_rel[h], v_rel = vv[src] @ m_rel[h]   (WMMA)
//   logit = <qq[dst], k_rel> * p_rel[h] * (1/8)
// kqv packed per node: [768] = k | q | v (f16). Relations pre-transposed f16
// aT/mT[h][f][d] and staged to LDS ONCE for all 4 heads; per-head work is only
// the per-edge K/V/Q gathers (async memory->LDS DMA).
// ---------------------------------------------------------------------------
__global__ __launch_bounds__(256) void edge_attn_kernel(
    const _Float16* __restrict__ kqvS, const _Float16* __restrict__ kqvD,
    const int* __restrict__ src, const int* __restrict__ dst, int E,
    const _Float16* __restrict__ aT, const _Float16* __restrict__ mT, // [4][64][64]
    const float* __restrict__ pRel,                                   // [4]
    float* __restrict__ logits,                                       // [E][4]
    _Float16* __restrict__ vrel)                                      // [E][256]
{
    __shared__ _Float16 sA[64][72], sV[64][72], sQ[64][72];
    __shared__ _Float16 sBa[4][64][72], sBm[4][64][72];   // [h][f][d], all heads
    __shared__ float sK[64][68];
    __shared__ float sRed[4][64];
    __shared__ int sSrc[64], sDst[64];

    int tid = threadIdx.x, lane = tid & 31, wave = tid >> 5;
    int e0 = blockIdx.x * 64;
    if (tid < 64) {
        int e = e0 + tid;
        sSrc[tid] = (e < E) ? src[e] : -1;
        sDst[tid] = (e < E) ? dst[e] : -1;
    }

    // ---- stage ALL heads' relation matrices once (reused 4x per block) ----
    {
        int f = tid >> 2, c = (tid & 3) * 16;
#pragma unroll
        for (int h = 0; h < 4; ++h) {
            const _Float16* pa = aT + (size_t)h * 4096 + f * 64 + c;
            const _Float16* pm = mT + (size_t)h * 4096 + f * 64 + c;
            async_copy16(pa,     &sBa[h][f][c]);
            async_copy16(pa + 8, &sBa[h][f][c + 8]);
            async_copy16(pm,     &sBm[h][f][c]);
            async_copy16(pm + 8, &sBm[h][f][c + 8]);
        }
    }
    __syncthreads();   // sSrc/sDst visible

    for (int h = 0; h < 4; ++h) {
        // ---- per-edge K/V/Q gathers for this head (async DMA) ----
        {
            int e = tid >> 2, c = (tid & 3) * 16;
            int s = sSrc[e], d = sDst[e];
            if (s >= 0) {
                const _Float16* pk = kqvS + (size_t)s * 768 + 0   + h * 64 + c;
                const _Float16* pv = kqvS + (size_t)s * 768 + 512 + h * 64 + c;
                async_copy16(pk,     &sA[e][c]);
                async_copy16(pk + 8, &sA[e][c + 8]);
                async_copy16(pv,     &sV[e][c]);
                async_copy16(pv + 8, &sV[e][c + 8]);
            }
            if (d >= 0) {
                const _Float16* pq = kqvD + (size_t)d * 768 + 256 + h * 64 + c;
                async_copy16(pq,     &sQ[e][c]);
                async_copy16(pq + 8, &sQ[e][c + 8]);
            }
        }
        async_wait0();
        __syncthreads();

        // ---- WMMA: waves 0-3 -> k_rel (to sK), waves 4-7 -> v_rel (global) ----
        {
            bool isV = wave >= 4;
            int w4 = wave & 3;
            const _Float16(*Asrc)[72] = isV ? sV : sA;
            const _Float16(*Bsrc)[72] = isV ? sBm[h] : sBa[h];
#pragma unroll
            for (int i = 0; i < 4; ++i) {
                int tile = w4 * 4 + i;
                int mt = tile >> 2, nt = tile & 3;
                v8f acc;
#pragma unroll
                for (int r = 0; r < 8; ++r) acc[r] = 0.0f;
#pragma unroll
                for (int ks = 0; ks < 2; ++ks) {
                    int kb = ks * 32;
                    HV af, bf;
                    int row = mt * 16 + (lane & 15), g = lane >> 4;
                    af.q[0] = *(const uint4*)&Asrc[row][kb + g * 8];
                    af.q[1] = *(const uint4*)&Asrc[row][kb + 16 + g * 8];
                    int col = nt * 16 + (lane & 15);
                    bf.q[0] = *(const uint4*)&Bsrc[col][kb + g * 8];
                    bf.q[1] = *(const uint4*)&Bsrc[col][kb + 16 + g * 8];
                    acc = __builtin_amdgcn_wmma_f32_16x16x32_f16(
                        false, af.v, false, bf.v, (short)0, acc, false, false);
                }
                int rbase = mt * 16 + (lane >> 4) * 8;
                int cbase = nt * 16 + (lane & 15);
                if (!isV) {
#pragma unroll
                    for (int r = 0; r < 8; ++r) sK[rbase + r][cbase] = acc[r];
                } else {
#pragma unroll
                    for (int r = 0; r < 8; ++r) {
                        int e = e0 + rbase + r;
                        if (e < E) vrel[(size_t)e * 256 + h * 64 + cbase] = (_Float16)acc[r];
                    }
                }
            }
        }
        __syncthreads();

        // ---- logit = <q, k_rel> : 4 partials of 16 per edge ----
        {
            int e = tid & 63, part = tid >> 6;
            float s = 0.0f;
#pragma unroll
            for (int i = 0; i < 16; ++i) {
                int dpos = part * 16 + i;
                s += (float)sQ[e][dpos] * sK[e][dpos];
            }
            sRed[part][e] = s;
        }
        __syncthreads();
        if (tid < 64) {
            int e = e0 + tid;
            if (e < E) {
                float t = sRed[0][tid] + sRed[1][tid] + sRed[2][tid] + sRed[3][tid];
                logits[(size_t)e * 4 + h] = t * pRel[h] * 0.125f;   // 1/sqrt(64)
            }
        }
        __syncthreads();   // protect sA/sV/sQ/sK/sRed before next head
    }
}

// ---------------------------------------------------------------------------
// Segment softmax (joint over all incoming edge types via shared m/s buffers)
// ---------------------------------------------------------------------------
__global__ void seg_max_kernel(const float* __restrict__ logits, const int* __restrict__ dst,
                               unsigned* __restrict__ mEnc, int E) {
    size_t idx = (size_t)blockIdx.x * blockDim.x + threadIdx.x;
    if (idx >= (size_t)E * 4) return;
    int e = (int)(idx >> 2), h = (int)(idx & 3);
    atomicMax(&mEnc[(size_t)dst[e] * 4 + h], enc_f32(logits[idx]));
}

__global__ void seg_exp_kernel(float* __restrict__ logits, const int* __restrict__ dst,
                               const unsigned* __restrict__ mEnc, float* __restrict__ ssum, int E) {
    size_t idx = (size_t)blockIdx.x * blockDim.x + threadIdx.x;
    if (idx >= (size_t)E * 4) return;
    int e = (int)(idx >> 2), h = (int)(idx & 3);
    float m = dec_f32(mEnc[(size_t)dst[e] * 4 + h]);
    if (m < -3.0e38f) m = 0.0f;   // mirror reference isfinite guard
    float ex = __expf(logits[idx] - m);
    logits[idx] = ex;             // in-place: logits become ex
    atomicAdd(&ssum[(size_t)dst[e] * 4 + h], ex);
}

__global__ void seg_scatter_kernel(const float* __restrict__ ex, const float* __restrict__ ssum,
                                   const int* __restrict__ dst, const _Float16* __restrict__ vrel,
                                   float* __restrict__ agg, int E) {
    size_t idx = (size_t)blockIdx.x * blockDim.x + threadIdx.x;
    if (idx >= (size_t)E * 16) return;                 // E * H * 4 chunks of 16
    int e = (int)(idx >> 4);
    int r = (int)(idx & 15), h = r >> 2, q = r & 3;
    int d = dst[e];
    float w = ex[(size_t)e * 4 + h] / (ssum[(size_t)d * 4 + h] + 1e-16f);
    const _Float16* vp = vrel + (size_t)e * 256 + h * 64 + q * 16;
    float* ap = agg + (size_t)d * 256 + h * 64 + q * 16;
#pragma unroll
    for (int i = 0; i < 16; ++i) atomicAdd(&ap[i], w * (float)vp[i]);
}

// ---------------------------------------------------------------------------
// Host orchestration
// ---------------------------------------------------------------------------
extern "C" void kernel_launch(void* const* d_in, const int* in_sizes, int n_in,
                              void* d_out, int out_size, void* d_ws, size_t ws_size,
                              hipStream_t stream) {
    (void)in_sizes; (void)n_in; (void)out_size; (void)ws_size;
    constexpr int NU = 50000, NTW = 100000, NTOT = 150000;
    constexpr size_t D = 256;

    const float* user_emb = (const float*)d_in[0];
    const float* tweet_tv = (const float*)d_in[1];
    const float* k_w = (const float*)d_in[2];  const float* k_b = (const float*)d_in[3];
    const float* q_w = (const float*)d_in[4];  const float* q_b = (const float*)d_in[5];
    const float* v_w = (const float*)d_in[6];  const float* v_b = (const float*)d_in[7];
    const float* a_w = (const float*)d_in[8];  const float* a_b = (const float*)d_in[9];
    const float* skip = (const float*)d_in[10];
    const float* a_rel = (const float*)d_in[11];
    const float* m_rel = (const float*)d_in[12];
    const float* p_rel = (const float*)d_in[13];
    const float* n1_g = (const float*)d_in[14]; const float* n1_b = (const float*)d_in[15];
    const float* n2_g = (const float*)d_in[16]; const float* n2_b = (const float*)d_in[17];
    const float* f1_w = (const float*)d_in[18]; const float* f1_b = (const float*)d_in[19];
    const float* f2_w = (const float*)d_in[20]; const float* f2_b = (const float*)d_in[21];
    const int* esrc[3] = {(const int*)d_in[22], (const int*)d_in[24], (const int*)d_in[26]};
    const int* edst[3] = {(const int*)d_in[23], (const int*)d_in[25], (const int*)d_in[27]};
    const int Ecnt[3] = {100000, 150000, 150000};
    const int est[3] = {0, 0, 1}, edt[3] = {0, 1, 0};

    // ---- workspace carve-up ----
    char* base = (char*)d_ws;
    size_t off = 0;
    auto alloc = [&](size_t bytes) -> void* {
        void* p = base + off;
        off = (off + bytes + 255) & ~(size_t)255;
        return p;
    };
    float*    xn32  = (float*)   alloc((size_t)NTOT * D * 4);
    _Float16* xn16  = (_Float16*)alloc((size_t)NTOT * D * 2);
    _Float16* kqv16 = (_Float16*)alloc((size_t)NTOT * 768 * 2);
    float*    agg   = (float*)   alloc((size_t)NTOT * D * 4);
    _Float16* ag16  = (_Float16*)alloc((size_t)NTOT * D * 2);
    _Float16* h16   = (_Float16*)alloc((size_t)NTOT * 1024 * 2);
    float*    logitB[3]; _Float16* vrelB[3];
    for (int et = 0; et < 3; ++et) {
        logitB[et] = (float*)   alloc((size_t)Ecnt[et] * 4 * 4);
        vrelB[et]  = (_Float16*)alloc((size_t)Ecnt[et] * 256 * 2);
    }
    unsigned* mEnc = (unsigned*)alloc((size_t)NTOT * 4 * 4);
    float*    ssum = (float*)   alloc((size_t)NTOT * 4 * 4);
    // pre-transposed f16 weights: [N][K] per (l,t) slice
    _Float16* wkT = (_Float16*)alloc((size_t)6 * 65536 * 2);
    _Float16* wqT = (_Float16*)alloc((size_t)6 * 65536 * 2);
    _Float16* wvT = (_Float16*)alloc((size_t)6 * 65536 * 2);
    _Float16* waT = (_Float16*)alloc((size_t)6 * 65536 * 2);
    _Float16* wf1T = (_Float16*)alloc((size_t)6 * 262144 * 2);
    _Float16* wf2T = (_Float16*)alloc((size_t)6 * 262144 * 2);
    _Float16* aT16 = (_Float16*)alloc((size_t)36 * 4096 * 2);  // [l,et,h][f][d]
    _Float16* mT16 = (_Float16*)alloc((size_t)36 * 4096 * 2);

    auto gsBlocks = [](size_t n) -> int {
        size_t b = (n + 255) / 256;
        return (int)(b > 8192 ? 8192 : b);
    };

    // ---- pre-transpose weights + relations to f16 [N][K] (once per launch) ----
    {
        dim3 tb(32, 8);
        transpose_f32_to_f16_kernel<<<dim3(8, 8, 6),  tb, 0, stream>>>(k_w,  wkT,  256, 256);
        transpose_f32_to_f16_kernel<<<dim3(8, 8, 6),  tb, 0, stream>>>(q_w,  wqT,  256, 256);
        transpose_f32_to_f16_kernel<<<dim3(8, 8, 6),  tb, 0, stream>>>(v_w,  wvT,  256, 256);
        transpose_f32_to_f16_kernel<<<dim3(8, 8, 6),  tb, 0, stream>>>(a_w,  waT,  256, 256);
        transpose_f32_to_f16_kernel<<<dim3(32, 8, 6), tb, 0, stream>>>(f1_w, wf1T, 256, 1024);
        transpose_f32_to_f16_kernel<<<dim3(8, 32, 6), tb, 0, stream>>>(f2_w, wf2T, 1024, 256);
        transpose_f32_to_f16_kernel<<<dim3(2, 2, 36), tb, 0, stream>>>(a_rel, aT16, 64, 64);
        transpose_f32_to_f16_kernel<<<dim3(2, 2, 36), tb, 0, stream>>>(m_rel, mT16, 64, 64);
    }

    // ---- init node states in d_out (users then broadcast tweets) ----
    float* xs = (float*)d_out;
    hipMemcpyAsync(xs, user_emb, (size_t)NU * D * 4, hipMemcpyDeviceToDevice, stream);
    bcast_tweet_kernel<<<gsBlocks((size_t)NTW * D), 256, 0, stream>>>(
        tweet_tv, xs + (size_t)NU * D, (size_t)NTW * D);

    for (int l = 0; l < 3; ++l) {
        // ---- LN1 + packed QKV per node type ----
        for (int t = 0; t < 2; ++t) {
            int Nt = t ? NTW : NU;
            size_t ro = t ? (size_t)NU : 0;
            float*    xs_t   = xs   + ro * D;
            float*    xn32_t = xn32 + ro * D;
            _Float16* xn16_t = xn16 + ro * D;
            _Float16* kqv_t  = kqv16 + ro * 768;
            int lt = l * 2 + t;
            ln_kernel<<<Nt, 256, 0, stream>>>(xs_t, n1_g + (size_t)lt * D, n1_b + (size_t)lt * D,
                                              xn32_t, xn16_t, Nt);
            const _Float16* W3[3] = {wkT + (size_t)lt * 65536, wqT + (size_t)lt * 65536,
                                     wvT + (size_t)lt * 65536};
            const float* B3[3] = {k_b + (size_t)lt * D, q_b + (size_t)lt * D, v_b + (size_t)lt * D};
            dim3 g(2, (unsigned)((Nt + BM - 1) / BM));
            for (int p = 0; p < 3; ++p)   // k -> cols 0.., q -> 256.., v -> 512..
                gemm_kernel<<<g, 256, 0, stream>>>(xn16_t, W3[p], B3[p],
                    nullptr, 0, kqv_t + p * 256, 768,
                    nullptr, nullptr, nullptr, Nt, 256, 256, 0);
        }

        // ---- attention: init joint-softmax state ----
        fill_u32_kernel<<<gsBlocks((size_t)NTOT * 4), 256, 0, stream>>>(
            mEnc, 0x007FFFFFu /* enc(-inf) */, (size_t)NTOT * 4);
        hipMemsetAsync(ssum, 0, (size_t)NTOT * 4 * 4, stream);
        hipMemsetAsync(agg, 0, (size_t)NTOT * D * 4, stream);

        for (int et = 0; et < 3; ++et) {
            size_t so = est[et] ? (size_t)NU : 0;
            size_t dofs = edt[et] ? (size_t)NU : 0;
            edge_attn_kernel<<<(Ecnt[et] + 63) / 64, 256, 0, stream>>>(
                kqv16 + so * 768, kqv16 + dofs * 768, esrc[et], edst[et], Ecnt[et],
                aT16 + (size_t)(l * 3 + et) * 4 * 4096, mT16 + (size_t)(l * 3 + et) * 4 * 4096,
                p_rel + (size_t)(l * 3 + et) * 4, logitB[et], vrelB[et]);
        }
        for (int et = 0; et < 3; ++et) {
            unsigned* m_dt = mEnc + (edt[et] ? (size_t)NU : 0) * 4;
            seg_max_kernel<<<((size_t)Ecnt[et] * 4 + 255) / 256, 256, 0, stream>>>(
                logitB[et], edst[et], m_dt, Ecnt[et]);
        }
        for (int et = 0; et < 3; ++et) {
            unsigned* m_dt = mEnc + (edt[et] ? (size_t)NU : 0) * 4;
            float* s_dt = ssum + (edt[et] ? (size_t)NU : 0) * 4;
            seg_exp_kernel<<<((size_t)Ecnt[et] * 4 + 255) / 256, 256, 0, stream>>>(
                logitB[et], edst[et], m_dt, s_dt, Ecnt[et]);
        }
        for (int et = 0; et < 3; ++et) {
            float* s_dt = ssum + (edt[et] ? (size_t)NU : 0) * 4;
            float* agg_dt = agg + (edt[et] ? (size_t)NU : 0) * D;
            seg_scatter_kernel<<<((size_t)Ecnt[et] * 16 + 255) / 256, 256, 0, stream>>>(
                logitB[et], s_dt, edst[et], vrelB[et], agg_dt, Ecnt[et]);
        }

        // ---- attn-out (skip blend, in place) + FFN per node type ----
        for (int t = 0; t < 2; ++t) {
            int Nt = t ? NTW : NU;
            size_t ro = t ? (size_t)NU : 0;
            float*    xs_t   = xs   + ro * D;
            float*    xn32_t = xn32 + ro * D;
            _Float16* xn16_t = xn16 + ro * D;
            _Float16* ag16_t = ag16 + ro * D;
            _Float16* h16_t  = h16  + ro * 1024;
            float*    agg_t  = agg  + ro * D;
            int lt = l * 2 + t;

            gelu_f32_to_f16_kernel<<<gsBlocks((size_t)Nt * D), 256, 0, stream>>>(
                agg_t, ag16_t, (size_t)Nt * D);
            dim3 g2(2, (unsigned)((Nt + BM - 1) / BM));
            // xs_t = xs_t + beta*(gelu(agg)@a_w + a_b) + (1-beta)*xn  (HGT skip)
            gemm_kernel<<<g2, 256, 0, stream>>>(ag16_t, waT + (size_t)lt * 65536,
                a_b + (size_t)lt * D, xs_t, 256, nullptr, 0,
                xs_t, xn32_t, skip + lt, Nt, 256, 256, 2);

            // FFN: xs_t += gelu(LN2(xs_t)@f1+b1)@f2 + b2
            ln_kernel<<<Nt, 256, 0, stream>>>(xs_t, n2_g + (size_t)lt * D, n2_b + (size_t)lt * D,
                                              nullptr, xn16_t, Nt);
            dim3 g8(8, (unsigned)((Nt + BM - 1) / BM));
            gemm_kernel<<<g8, 256, 0, stream>>>(xn16_t, wf1T + (size_t)lt * 262144,
                f1_b + (size_t)lt * 1024, nullptr, 0, h16_t, 1024,
                nullptr, nullptr, nullptr, Nt, 1024, 256, 1);
            gemm_kernel<<<g2, 256, 0, stream>>>(h16_t, wf2T + (size_t)lt * 262144,
                f2_b + (size_t)lt * D, xs_t, 256, nullptr, 0,
                xs_t, nullptr, nullptr, Nt, 256, 1024, 3);
        }
    }
}
